// MultiHeadSelfAttention_31001073943138
// MI455X (gfx1250) — matrix operand order repaired and checked by reading, not compiled, not run on hardware
//
#include <hip/hip_runtime.h>

// ---------------------------------------------------------------------------
// MI455X (gfx1250) multi-head self-attention, bf16 WMMA + f32 accumulate.
// B=2, S=2048, D_MODEL=2048, H=16, D_K=128. wave32 only.
// ---------------------------------------------------------------------------

typedef unsigned short u16;
typedef unsigned int   u32;

typedef __attribute__((ext_vector_type(16))) __bf16 v16bf;
typedef __attribute__((ext_vector_type(8)))  float  v8f;

struct __attribute__((aligned(16))) U128 { u32 a, b, c, d; };
union FragBF { U128 q[2]; v16bf v; };

#define SEQL   2048
#define DMODEL 2048
#define HEADS  16
#define DK     128

// log2(e) / sqrt(128): folded into Q at RoPE time so flash softmax runs in
// base-2 with raw v_exp_f32 and no per-score scaling multiplies.
#define QSCALE 0.12751589167446215f

// ----------------------------- helpers -------------------------------------

__device__ __forceinline__ u16 f2bf(float f) {
  u32 u = __float_as_uint(f);
  u32 r = u + 0x7FFFu + ((u >> 16) & 1u);   // round-to-nearest-even
  return (u16)(r >> 16);
}
__device__ __forceinline__ float bf2f(u16 h) {
  return __uint_as_float(((u32)h) << 16);
}
__device__ __forceinline__ v8f v8f_zero() {
  v8f z;
#pragma unroll
  for (int i = 0; i < 8; ++i) z[i] = 0.f;
  return z;
}
__device__ __forceinline__ v8f wmma_bf16(v16bf a, v16bf b, v8f c) {
  // v_wmma_f32_16x16x32_bf16 : D = A(16x32) * B(32x16) + C(16x16 f32)
  return __builtin_amdgcn_wmma_f32_16x16x32_bf16(
      false, a, false, b, (short)0, c, false, false);
}

// A fragment (16x32 bf16, M x K): lane l -> row m0+(l&15),
// elems 0..7  = K[k0 + (l>>4)*8 .. +7], elems 8..15 = K[k0+16+(l>>4)*8 .. +7]
__device__ __forceinline__ v16bf load_frag_a(const u16* __restrict__ A, int lda,
                                             int m0, int k0, int lane) {
  const int row = m0 + (lane & 15);
  const int kb  = k0 + ((lane >> 4) << 3);
  FragBF f;
  f.q[0] = *(const U128*)(A + (size_t)row * lda + kb);
  f.q[1] = *(const U128*)(A + (size_t)row * lda + kb + 16);
  return f.v;
}

// B fragment (32x16 bf16, K x N) for NT GEMM: B[k][n] = Bp[n][k] row-major.
// lane l -> col n0+(l&15), elems 0..7 = K[k0+(l>>4)*8 .. +7], elems 8..15 = K[k0+16+(l>>4)*8 .. +7].
__device__ __forceinline__ v16bf load_frag_b(const u16* __restrict__ Bp, int ldb,
                                             int n0, int k0, int lane) {
  const int row = n0 + (lane & 15);
  const int kb  = k0 + ((lane >> 4) << 3);
  FragBF f;
  f.q[0] = *(const U128*)(Bp + (size_t)row * ldb + kb);
  f.q[1] = *(const U128*)(Bp + (size_t)row * ldb + kb + 16);
  return f.v;
}

// ----------------------- f32 -> bf16 conversion ----------------------------

__global__ void cvt_f32_bf16_kernel(const float* __restrict__ in,
                                    u16* __restrict__ out, int n) {
  int i = blockIdx.x * blockDim.x + threadIdx.x;
  if (i < n) out[i] = f2bf(in[i]);
}

// ------------------------------ NT GEMM ------------------------------------
// C[M,N] = A[M,K] * B[N,K]^T   (both row-major bf16, f32 accumulate)
// 256 threads = 8 waves (4x2); 32x64 tile per wave; 128x128 per block.
// 8 WMMAs per 6 fragment loads per k-step of 32; k-loop unrolled x2 so the
// scheduler can overlap next-step loads with current WMMAs.
// OUT_MODE: 0 = bf16 store, 1 = f32 store.
template <int OUT_MODE>
__global__ __launch_bounds__(256) void gemm_nt_kernel(
    const u16* __restrict__ A, const u16* __restrict__ B,
    void* __restrict__ C, int M, int N, int K) {
  const int tid  = threadIdx.x;
  const int w    = tid >> 5;
  const int lane = tid & 31;
  const int wm   = w >> 1;                  // 0..3
  const int wn   = w & 1;                   // 0..1
  const int m0   = blockIdx.y * 128 + wm * 32;
  const int n0   = blockIdx.x * 128 + wn * 64;

  v8f acc[2][4];
#pragma unroll
  for (int i = 0; i < 2; ++i)
#pragma unroll
    for (int j = 0; j < 4; ++j) acc[i][j] = v8f_zero();

#pragma unroll 2
  for (int k0 = 0; k0 < K; k0 += 32) {
    v16bf a0 = load_frag_a(A, K, m0,      k0, lane);
    v16bf a1 = load_frag_a(A, K, m0 + 16, k0, lane);
    v16bf bf[4];
#pragma unroll
    for (int j = 0; j < 4; ++j)
      bf[j] = load_frag_b(B, K, n0 + 16 * j, k0, lane);
#pragma unroll
    for (int j = 0; j < 4; ++j) {
      acc[0][j] = wmma_bf16(a0, bf[j], acc[0][j]);
      acc[1][j] = wmma_bf16(a1, bf[j], acc[1][j]);
    }
  }

  const int hf = lane >> 4, ln = lane & 15;
#pragma unroll
  for (int i = 0; i < 2; ++i)
#pragma unroll
    for (int j = 0; j < 4; ++j)
#pragma unroll
      for (int r = 0; r < 8; ++r) {
        const int row = m0 + i * 16 + r + 8 * hf;
        const int col = n0 + j * 16 + ln;
        if (OUT_MODE == 0)
          ((u16*)C)[(size_t)row * N + col] = f2bf(acc[i][j][r]);
        else
          ((float*)C)[(size_t)row * N + col] = acc[i][j][r];
      }
}

// ---------------------- RoPE (in-place) + V transpose ----------------------
// One thread per interleaved pair (d=2j, 2j+1). Q additionally absorbs the
// softmax scale (log2e/sqrt(d_k)). Also scatters V into Vt[b][h][d][s] so PV
// WMMA B-fragments are contiguous.
__global__ void rope_vt_kernel(u16* __restrict__ Qb, u16* __restrict__ Kb,
                               const u16* __restrict__ Vb, u16* __restrict__ Vt,
                               const int* __restrict__ pos, int npairs) {
  int i = blockIdx.x * blockDim.x + threadIdx.x;
  if (i >= npairs) return;
  const int j = i & 63;           // half-dim index 0..63
  const int h = (i >> 6) & 15;
  const int s = (i >> 10) & 2047;
  const int b = i >> 21;

  const float p    = (float)pos[b * SEQL + s];
  // theta^(-2j/128) = exp(-j * ln(10000)/64)
  const float freq = __expf(-0.14391156831212787f * (float)j);
  float sn, cs;
  __sincosf(p * freq, &sn, &cs);

  const size_t idx = ((size_t)(b * SEQL + s)) * DMODEL + h * DK + 2 * j;

  float x1 = bf2f(Qb[idx]), x2 = bf2f(Qb[idx + 1]);
  Qb[idx]     = f2bf((x1 * cs - x2 * sn) * QSCALE);
  Qb[idx + 1] = f2bf((x1 * sn + x2 * cs) * QSCALE);

  x1 = bf2f(Kb[idx]); x2 = bf2f(Kb[idx + 1]);
  Kb[idx]     = f2bf(x1 * cs - x2 * sn);
  Kb[idx + 1] = f2bf(x1 * sn + x2 * cs);

  const size_t vt = ((size_t)(b * HEADS + h) * DK + 2 * j) * SEQL + s;
  Vt[vt]        = Vb[idx];
  Vt[vt + SEQL] = Vb[idx + 1];
}

// --------------------------- flash attention -------------------------------
// Grid: (S/64 q-tiles, B*H). 128 threads = 4 waves; each wave owns 16 q rows.
// Base-2 online softmax over 32-wide key chunks (scale pre-folded into Q);
// causal mask is arithmetic so EXEC stays all-ones for WMMA. P is relaid out
// C-frag -> A-frag via per-wave LDS. All register-array loops fully unrolled.
__global__ __launch_bounds__(128) void flash_attn_kernel(
    const u16* __restrict__ Qb, const u16* __restrict__ Kb,
    const u16* __restrict__ Vt, u16* __restrict__ Ob) {
  __shared__ __align__(16) u16 lds_p[4][16 * 32];

  const int tid  = threadIdx.x;
  const int w    = tid >> 5;
  const int lane = tid & 31;
  const int hf   = lane >> 4;
  const int ln   = lane & 15;
  const int bh   = blockIdx.y;
  const int b    = bh >> 4;
  const int h    = bh & 15;
  const int qbase = blockIdx.x * 64 + w * 16;
  const int qlast = qbase + 15;

  const u16* qp = Qb + (size_t)b * SEQL * DMODEL;
  const u16* kp = Kb + (size_t)b * SEQL * DMODEL;
  const u16* vp = Vt + (size_t)bh * DK * SEQL;

  // Q fragments for all 4 K-chunks of d_k=128, loaded once.
  v16bf aq[4];
#pragma unroll
  for (int c = 0; c < 4; ++c)
    aq[c] = load_frag_a(qp, DMODEL, qbase, h * DK + 32 * c, lane);

  v8f av[8];
#pragma unroll
  for (int j = 0; j < 8; ++j) av[j] = v8f_zero();
  float mrun[8], lrun[8];
#pragma unroll
  for (int r = 0; r < 8; ++r) { mrun[r] = -3.0e38f; lrun[r] = 0.f; }

  u16* lp = &lds_p[w][0];

  for (int kc = 0; kc <= qlast; kc += 32) {
    // scores (already in log2 domain): 16 q-rows x 32 keys, f32 accum
    v8f s0 = v8f_zero(), s1 = v8f_zero();
#pragma unroll
    for (int c = 0; c < 4; ++c) {
      v16bf bk0 = load_frag_b(kp, DMODEL, kc,      h * DK + 32 * c, lane);
      v16bf bk1 = load_frag_b(kp, DMODEL, kc + 16, h * DK + 32 * c, lane);
      s0 = wmma_bf16(aq[c], bk0, s0);
      s1 = wmma_bf16(aq[c], bk1, s1);
    }

    // base-2 online softmax over this 32-wide chunk (fully unrolled)
#pragma unroll
    for (int r = 0; r < 8; ++r) {
      const int row = qbase + r + 8 * hf;
      const int c0  = kc + ln;
      const int c1  = kc + 16 + ln;
      float x0 = (c0 <= row) ? s0[r] : -3.0e38f;
      float x1 = (c1 <= row) ? s1[r] : -3.0e38f;
      float mx = fmaxf(x0, x1);
#pragma unroll
      for (int off = 1; off < 16; off <<= 1)
        mx = fmaxf(mx, __shfl_xor(mx, off, 32));
      const float mnew  = fmaxf(mrun[r], mx);
      const float p0    = __builtin_amdgcn_exp2f(x0 - mnew);
      const float p1    = __builtin_amdgcn_exp2f(x1 - mnew);
      const float alpha = __builtin_amdgcn_exp2f(mrun[r] - mnew);
      mrun[r] = mnew;
      float ps = p0 + p1;
#pragma unroll
      for (int off = 1; off < 16; off <<= 1)
        ps += __shfl_xor(ps, off, 32);
      lrun[r] = lrun[r] * alpha + ps;
#pragma unroll
      for (int j = 0; j < 8; ++j) av[j][r] *= alpha;
      const int lrow = r + 8 * hf;                 // C-frag row in tile
      lp[lrow * 32 + ln]      = f2bf(p0);
      lp[lrow * 32 + 16 + ln] = f2bf(p1);
    }

    // C-frag -> A-frag relayout through per-wave LDS (same-wave DS ops are
    // in-order; DScnt wait inserted by compiler before the b128 reads).
    FragBF pf;
    pf.q[0] = *(const U128*)(lp + ln * 32 + hf * 8);
    pf.q[1] = *(const U128*)(lp + ln * 32 + 16 + hf * 8);

    // O += P(16x32) * V(32x128)
#pragma unroll
    for (int j = 0; j < 8; ++j) {
      v16bf bv = load_frag_b(vp, SEQL, j * 16, kc, lane);
      av[j] = wmma_bf16(pf.v, bv, av[j]);
    }
  }

  // normalize + store bf16 O in [b, s, h*128+d] layout (A-operand of out-proj)
#pragma unroll
  for (int r = 0; r < 8; ++r) {
    const int   row = qbase + r + 8 * hf;
    const float inv = 1.0f / lrun[r];
    const size_t base = ((size_t)b * SEQL + row) * DMODEL + h * DK;
#pragma unroll
    for (int j = 0; j < 8; ++j)
      Ob[base + j * 16 + ln] = f2bf(av[j][r] * inv);
  }
}

// ------------------------------- launch ------------------------------------

extern "C" void kernel_launch(void* const* d_in, const int* in_sizes, int n_in,
                              void* d_out, int out_size, void* d_ws, size_t ws_size,
                              hipStream_t stream) {
  (void)in_sizes; (void)n_in; (void)out_size; (void)ws_size;

  const float* x   = (const float*)d_in[0];
  const int*   pos = (const int*)d_in[1];
  const float* Wq  = (const float*)d_in[2];
  const float* Wk  = (const float*)d_in[3];
  const float* Wv  = (const float*)d_in[4];
  const float* Wo  = (const float*)d_in[5];
  float* out = (float*)d_out;

  const size_t XN = (size_t)2 * SEQL * DMODEL;      // 8,388,608
  const size_t WN = (size_t)DMODEL * DMODEL;        // 4,194,304

  u16* ws16 = (u16*)d_ws;                           // 128 MB total
  u16* xb  = ws16;                                  // x bf16
  u16* wqb = xb  + XN;
  u16* wkb = wqb + WN;
  u16* wvb = wkb + WN;
  u16* wob = wvb + WN;
  u16* qb  = wob + WN;                              // Q bf16 (RoPE'd + scaled)
  u16* kb  = qb  + XN;
  u16* vb  = kb  + XN;
  u16* vt  = vb  + XN;                              // V transposed [b,h,d,s]
  u16* ob  = vt  + XN;                              // attention output bf16

  const int M = 2 * SEQL, N = DMODEL, K = DMODEL;

  // 1) bf16 conversions
  cvt_f32_bf16_kernel<<<(int)((XN + 255) / 256), 256, 0, stream>>>(x, xb, (int)XN);
  cvt_f32_bf16_kernel<<<(int)((WN + 255) / 256), 256, 0, stream>>>(Wq, wqb, (int)WN);
  cvt_f32_bf16_kernel<<<(int)((WN + 255) / 256), 256, 0, stream>>>(Wk, wkb, (int)WN);
  cvt_f32_bf16_kernel<<<(int)((WN + 255) / 256), 256, 0, stream>>>(Wv, wvb, (int)WN);
  cvt_f32_bf16_kernel<<<(int)((WN + 255) / 256), 256, 0, stream>>>(Wo, wob, (int)WN);

  // 2) QKV projections (NT GEMM, bf16 out)
  dim3 gg(N / 128, M / 128);
  gemm_nt_kernel<0><<<gg, 256, 0, stream>>>(xb, wqb, qb, M, N, K);
  gemm_nt_kernel<0><<<gg, 256, 0, stream>>>(xb, wkb, kb, M, N, K);
  gemm_nt_kernel<0><<<gg, 256, 0, stream>>>(xb, wvb, vb, M, N, K);

  // 3) RoPE on Q,K (in place, Q absorbs softmax scale) + V transpose
  const int npairs = (int)(XN / 2);
  rope_vt_kernel<<<(npairs + 255) / 256, 256, 0, stream>>>(qb, kb, vb, vt, pos, npairs);

  // 4) causal flash attention
  dim3 ga(SEQL / 64, 2 * HEADS);
  flash_attn_kernel<<<ga, 128, 0, stream>>>(qb, kb, vt, ob);

  // 5) output projection (f32 out -> d_out)
  gemm_nt_kernel<1><<<gg, 256, 0, stream>>>(ob, wob, out, M, N, K);
}
